// GALinearAttentionModel_36232344109553
// MI455X (gfx1250) — compile-verified, matrix operand order checked by reading, not run on hardware
//
#include <hip/hip_runtime.h>
#include <hip/hip_bf16.h>

typedef __attribute__((ext_vector_type(16))) _Float16 v16h;
typedef __attribute__((ext_vector_type(8)))  float    v8f;

#define LOOKBACK 8
#define HID      32
#define MV       11
#define TILE_Q   128                 // queries per block
#define NPOS     (TILE_Q + 16)       // 144 positions (128 + 8 halo, rounded to 16)
#define NTILES   (NPOS / 16)         // 9 WMMA row-tiles
#define PITCH    33                  // LDS row pitch in floats (bank-conflict-free)
#define EPSV     1e-6f

__device__ __forceinline__ float phi_f(float x) {
    return (x >= 0.0f ? x : 0.01f * x) + 1.0f;
}

// K-slot index for 16-bit A/B fragment element (vgpr v, element e, lane-half hk)
__device__ __forceinline__ int kslot(int v, int e, int hk) {
    return (v < 4) ? (hk * 8 + 2 * v + e) : (16 + hk * 8 + 2 * (v - 4) + e);
}

__global__ __launch_bounds__(128)
void ga_attn_wmma_kernel(const float* __restrict__ x,
                         const float* __restrict__ scalar_bias,
                         const float* __restrict__ vscale,
                         const float* __restrict__ vbias,
                         const float* __restrict__ bw,
                         const float* __restrict__ WQ,
                         const float* __restrict__ WK,
                         const float* __restrict__ WV,
                         const float* __restrict__ w_head,
                         const float* __restrict__ b_head,
                         float* __restrict__ out_pred,
                         float* __restrict__ out_w,
                         int L) {
    __shared__ float sQ[NPOS * PITCH];
    __shared__ float sK[NPOS * PITCH];
    __shared__ float sV[NPOS * PITCH];

    const int b    = blockIdx.y;
    const int T    = L - LOOKBACK;
    const int p0   = LOOKBACK + blockIdx.x * TILE_Q;  // first query position
    const int base = p0 - LOOKBACK;                   // global position of LDS row 0

    const int tid  = threadIdx.x;
    const int wave = tid >> 5;
    const int lane = tid & 31;
    const int hk   = lane >> 4;      // K-range half (A/B fragments)
    const int ncol = lane & 15;      // column within 16-wide fragment

    // ---- B fragments: W_Q/W_K/W_V as 32x16 f16, two 16-column halves each ----
    v16h bq[2], bk[2], bv[2];
#pragma unroll
    for (int nh = 0; nh < 2; ++nh) {
        const int n = ncol + 16 * nh;
#pragma unroll
        for (int v = 0; v < 8; ++v) {
#pragma unroll
            for (int e = 0; e < 2; ++e) {
                const int k = kslot(v, e, hk);
                const bool ok = (k < MV);
                bq[nh][2 * v + e] = (_Float16)(ok ? WQ[k * HID + n] : 0.0f);
                bk[nh][2 * v + e] = (_Float16)(ok ? WK[k * HID + n] : 0.0f);
                bv[nh][2 * v + e] = (_Float16)(ok ? WV[k * HID + n] : 0.0f);
            }
        }
    }

    const float sb  = 1.0f + scalar_bias[0];
    const float vs0 = vscale[0], vs1 = vscale[1], vs2 = vscale[2], vs3 = vscale[3];
    const float vb0 = vbias[0],  vb1 = vbias[1],  vb2 = vbias[2],  vb3 = vbias[3];
    const float w0 = bw[0], w1 = bw[1], w2 = bw[2], w3 = bw[3], w4 = bw[4], w5 = bw[5];

    // ---- Projection phase: each wave owns tiles wave, wave+4, wave+8 (uniform) ----
    for (int slot = 0; slot < 3; ++slot) {
        const int tile = wave + 4 * slot;
        if (tile >= NTILES) break;                     // wave-uniform branch

        // Build A fragment (16x32 f16): both lane halves carry rows 0..15 of tile
        const int mrow = lane & 15;
        int pos = base + tile * 16 + mrow;
        if (pos > L - 1) pos = L - 1;                  // clamp (tail tiles)
        const float* xr = x + ((size_t)b * L + (size_t)pos) * 4;
        const float x0 = xr[0], x1 = xr[1], x2 = xr[2], x3 = xr[3];

        float Mv[MV];
        Mv[0]  = sb;
        Mv[1]  = x0 * vs0 + vb0;
        Mv[2]  = x1 * vs1 + vb1;
        Mv[3]  = x2 * vs2 + vb2;
        Mv[4]  = x3 * vs3 + vb3;
        Mv[5]  = w0 * (x0 - x1);
        Mv[6]  = w1 * (x0 - x2);
        Mv[7]  = w2 * (x0 - x3);
        Mv[8]  = w3 * (x1 - x2);
        Mv[9]  = w4 * (x1 - x3);
        Mv[10] = w5 * (x2 - x3);

        v16h a;
#pragma unroll
        for (int v = 0; v < 8; ++v) {
#pragma unroll
            for (int e = 0; e < 2; ++e) {
                const int k = kslot(v, e, hk);
                a[2 * v + e] = (k < MV) ? (_Float16)Mv[k] : (_Float16)0.0f;
            }
        }

        v8f dq0 = {}, dq1 = {}, dk0 = {}, dk1 = {}, dv0 = {}, dv1 = {};
        dq0 = __builtin_amdgcn_wmma_f32_16x16x32_f16(false, a, false, bq[0], (short)0, dq0, false, false);
        dq1 = __builtin_amdgcn_wmma_f32_16x16x32_f16(false, a, false, bq[1], (short)0, dq1, false, false);
        dk0 = __builtin_amdgcn_wmma_f32_16x16x32_f16(false, a, false, bk[0], (short)0, dk0, false, false);
        dk1 = __builtin_amdgcn_wmma_f32_16x16x32_f16(false, a, false, bk[1], (short)0, dk1, false, false);
        dv0 = __builtin_amdgcn_wmma_f32_16x16x32_f16(false, a, false, bv[0], (short)0, dv0, false, false);
        dv1 = __builtin_amdgcn_wmma_f32_16x16x32_f16(false, a, false, bv[1], (short)0, dv1, false, false);

        // Spill C/D to LDS: VGPR r holds row r (lanes 0-15) / row r+8 (lanes 16-31)
        const int rowBase = tile * 16 + hk * 8;
#pragma unroll
        for (int r = 0; r < 8; ++r) {
            const int row = rowBase + r;
            sQ[row * PITCH + ncol]      = phi_f(dq0[r]);
            sQ[row * PITCH + ncol + 16] = phi_f(dq1[r]);
            sK[row * PITCH + ncol]      = phi_f(dk0[r]);
            sK[row * PITCH + ncol + 16] = phi_f(dk1[r]);
            sV[row * PITCH + ncol]      = dv0[r];
            sV[row * PITCH + ncol + 16] = dv1[r];
        }
    }

    __syncthreads();

    // ---- Attention phase: one thread per query ----
    const int t = blockIdx.x * TILE_Q + tid;          // output time index
    if (t < T) {
        // query lives at LDS row tid + 8; window keys at rows tid .. tid+7
        float qreg[HID];
#pragma unroll
        for (int h = 0; h < HID; ++h) qreg[h] = sQ[(tid + LOOKBACK) * PITCH + h];

        float s[LOOKBACK];
        float den = EPSV;
#pragma unroll
        for (int l = 0; l < LOOKBACK; ++l) {
            const float* kr = &sK[(tid + l) * PITCH];
            float acc = 0.0f;
#pragma unroll
            for (int h = 0; h < HID; ++h) acc += qreg[h] * kr[h];
            s[l] = acc;
            den += acc;
        }
        const float inv = 1.0f / den;

        float pred = 0.0f;
#pragma unroll
        for (int g = 0; g < HID; ++g) {
            float num = 0.0f;
#pragma unroll
            for (int l = 0; l < LOOKBACK; ++l) num += s[l] * sV[(tid + l) * PITCH + g];
            pred += (num * inv) * w_head[g];
        }
        pred += b_head[0];

        const size_t bt = (size_t)b * T + (size_t)t;
        out_pred[bt] = pred;
        float* wout = out_w + bt * LOOKBACK;
#pragma unroll
        for (int l = 0; l < LOOKBACK; ++l) wout[l] = s[l] * inv;
    }
}

extern "C" void kernel_launch(void* const* d_in, const int* in_sizes, int n_in,
                              void* d_out, int out_size, void* d_ws, size_t ws_size,
                              hipStream_t stream) {
    const float* x   = (const float*)d_in[0];
    const float* sb  = (const float*)d_in[1];
    const float* vs  = (const float*)d_in[2];
    const float* vb  = (const float*)d_in[3];
    const float* bw  = (const float*)d_in[4];
    const float* WQ  = (const float*)d_in[5];
    const float* WK  = (const float*)d_in[6];
    const float* WV  = (const float*)d_in[7];
    const float* wh  = (const float*)d_in[8];
    const float* bh  = (const float*)d_in[9];

    const int B = 8;
    const int L = in_sizes[0] / (B * 4);
    const int T = L - LOOKBACK;

    float* out_pred = (float*)d_out;           // B*T predictions
    float* out_w    = out_pred + (size_t)B * T; // B*T*8 weights

    dim3 grid((T + TILE_Q - 1) / TILE_Q, B);
    dim3 block(TILE_Q);
    ga_attn_wmma_kernel<<<grid, block, 0, stream>>>(
        x, sb, vs, vb, bw, WQ, WK, WV, wh, bh, out_pred, out_w, L);
}
